// GptOssSparseMoeBlock_46059229283039
// MI455X (gfx1250) — matrix operand order repaired and checked by reading, not compiled, not run on hardware
//
#include <hip/hip_runtime.h>

// ---- problem constants ----
static constexpr int kT = 1024;
static constexpr int kH = 2880;
static constexpr int kI = 2880;
static constexpr int kE = 16;
static constexpr float kAlpha = 1.702f;
static constexpr float kLimit = 7.0f;
// sum_e pad16(cnt_e) <= 4096 + 16*15 = 4336; +64 rows slack for 64-row
// down-GEMM A-tile over-read past the last expert's padded count.
static constexpr int kRowsMax = 4400;

typedef __bf16 bf16;
typedef __attribute__((ext_vector_type(2)))  __bf16 v2bf;
typedef __attribute__((ext_vector_type(16))) __bf16 v16bf;
typedef __attribute__((ext_vector_type(8)))  float  v8f;
typedef __attribute__((ext_vector_type(4)))  unsigned v4u;
typedef __attribute__((ext_vector_type(8)))  int      v8i;
typedef __attribute__((ext_vector_type(4)))  int      v4i;

union V16B {
  uint4 q[2];
  v16bf v;
};

// pack two fp32 -> packed bf16 dword: exactly one v_cvt_pk_bf16_f32
__device__ __forceinline__ unsigned pk2(float a, float b) {
#if __has_builtin(__builtin_amdgcn_cvt_pk_bf16_f32)
  union { v2bf h; unsigned u; } x;
  x.h = __builtin_amdgcn_cvt_pk_bf16_f32(a, b);
  return x.u;
#else
  unsigned r;
  asm("v_cvt_pk_bf16_f32 %0, %1, %2" : "=v"(r) : "v"(a), "v"(b));
  return r;
#endif
}

// ---------------------------------------------------------------------------
// CDNA5 async global->LDS copy (ASYNCcnt-tracked). LDS address operand is the
// low 32 bits of the generic pointer (LDS aperture maps addr[31:0] -> offset).
// ---------------------------------------------------------------------------
__device__ __forceinline__ void async_copy_b128(unsigned lds_off, const void* g) {
  asm volatile("global_load_async_to_lds_b128 %0, %1, off"
               :: "v"(lds_off), "v"(g) : "memory");
}
__device__ __forceinline__ void wait_async() {
  asm volatile("s_wait_asynccnt 0x0" ::: "memory");
}
__device__ __forceinline__ void wait_tensor() {
#if __has_builtin(__builtin_amdgcn_s_wait_tensorcnt)
  __builtin_amdgcn_s_wait_tensorcnt(0);
#else
  asm volatile("s_wait_tensorcnt 0x0" ::: "memory");
#endif
}

#if __has_builtin(__builtin_amdgcn_tensor_load_to_lds)
#define USE_TDM 1
// TDM 2D tile load: tile_dim0 x tile_dim1 elements of data_size=2B starting
// at byte address `ga`, row stride `stride0` (elements), into LDS at lds_off.
// D# layout per CDNA5 ISA (group0 + group1).
// This toolchain's builtin takes 6 args: (g0, g1, g2, g3, g4, cpol).
__device__ __forceinline__ void tdm_load_2d(unsigned lds_off, unsigned long long ga,
                                            int stride0) {
  v4u g0;
  g0[0] = 1u;                                            // count=1, user mode
  g0[1] = lds_off;                                       // lds_addr
  g0[2] = (unsigned)ga;                                  // global_addr[31:0]
  g0[3] = (unsigned)((ga >> 32) & 0x01FFFFFFu) | 0x80000000u;  // [56:32]|type=2
  v8i g1;
  g1[0] = 0x00010000;             // workgroup_mask=0, data_size=1 (2 bytes)
  g1[1] = 0;                      // no atomic barrier; tensor_dim0[15:0]=0
  g1[2] = 0x4000;                 // tensor_dim0=0x40000000 (huge, no OOB)
  g1[3] = (32 << 16) | 0x4000;    // tensor_dim1=0x40000000; tile_dim0=32
  g1[4] = 64;                     // tile_dim1=64, tile_dim2=0
  g1[5] = stride0;                // tensor_dim0_stride[31:0]
  g1[6] = 0;                      // stride0 hi, dim1_stride lo
  g1[7] = 0;
  v4i z4 = {0, 0, 0, 0};
  v8i z8 = {0, 0, 0, 0, 0, 0, 0, 0};
  __builtin_amdgcn_tensor_load_to_lds(g0, g1, z4, z4, z8, 0);
}
#else
#define USE_TDM 0
#endif

// ---------------------------------------------------------------------------
// WMMA fragment loaders (wave32), layouts per CDNA5 ISA 7.12.2.
// A: 16x32 bf16 row-major tile (ld=32). lane0-15: M=lane, K=0..15 (contiguous k order, as frag_bt);
// lane16-31: M=lane-16, K=16..31. Two contiguous 16B reads -> ds_load_b128 x2.
// ---------------------------------------------------------------------------
__device__ __forceinline__ v16bf frag_a(const bf16* __restrict__ p) {
  int lane = threadIdx.x & 31;
  const bf16* r = p + (lane & 15) * 32 + ((lane >> 4) << 4);
  V16B f;
  f.q[0] = *(const uint4*)r;         // K = kb .. kb+7
  f.q[1] = *(const uint4*)(r + 8);   // K = kb+8 .. kb+15
  return f.v;
}

// B: 32x16 (KxN) tile stored TRANSPOSED in LDS as Wt[n][k] (32 K contiguous
// per N row). lane n = lane&15, K half = (lane>>4)*16; 16 contiguous bf16.
__device__ __forceinline__ v16bf frag_bt(const bf16* __restrict__ p) {
  int lane = threadIdx.x & 31;
  const bf16* r = p + (lane & 15) * 32 + ((lane >> 4) << 4);
  V16B f;
  f.q[0] = *(const uint4*)r;        // K = kb .. kb+7
  f.q[1] = *(const uint4*)(r + 8);  // K = kb+8 .. kb+15
  return f.v;
}

// ---------------------------------------------------------------------------
// small prep kernels
// ---------------------------------------------------------------------------
__global__ __launch_bounds__(32) void moe_init(int* __restrict__ cnt) {
  if (threadIdx.x < kE) cnt[threadIdx.x] = 0;
}

__global__ __launch_bounds__(256) void moe_cvt(const float* __restrict__ x,
                                               bf16* __restrict__ xb) {
  size_t i = ((size_t)blockIdx.x * 256 + threadIdx.x) * 4;
  float4 v = *(const float4*)(x + i);
  unsigned* dst = (unsigned*)(xb + i);
  dst[0] = pk2(v.x, v.y);
  dst[1] = pk2(v.z, v.w);
}

// one wave32 per token: 16 dot products over H, shuffle-reduced, then
// softmax + top-4 + renormalize in lane 0.
__global__ __launch_bounds__(256) void moe_router(const float* __restrict__ x,
                                                  const float* __restrict__ rw,
                                                  const float* __restrict__ rb,
                                                  float* __restrict__ wts,
                                                  int* __restrict__ eidx) {
  int lane = threadIdx.x & 31;
  int t = blockIdx.x * 8 + (threadIdx.x >> 5);
  const float* xr = x + (size_t)t * kH;
  float acc[kE];
#pragma unroll
  for (int e = 0; e < kE; ++e) acc[e] = 0.f;
  for (int h = lane; h < kH; h += 32) {
    float xv = xr[h];
#pragma unroll
    for (int e = 0; e < kE; ++e) acc[e] = fmaf(xv, rw[(size_t)e * kH + h], acc[e]);
  }
#pragma unroll
  for (int e = 0; e < kE; ++e) {
#pragma unroll
    for (int off = 16; off > 0; off >>= 1) acc[e] += __shfl_xor(acc[e], off, 32);
  }
  if (lane == 0) {
    float lg[kE], mx = -1e30f;
#pragma unroll
    for (int e = 0; e < kE; ++e) {
      lg[e] = acc[e] + rb[e];
      mx = fmaxf(mx, lg[e]);
    }
    float p[kE], sum = 0.f;
#pragma unroll
    for (int e = 0; e < kE; ++e) {
      p[e] = __expf(lg[e] - mx);
      sum += p[e];
    }
    float inv = __builtin_amdgcn_rcpf(sum);
#pragma unroll
    for (int e = 0; e < kE; ++e) p[e] *= inv;
    int te[4];
    float tv[4], tsum = 0.f;
    unsigned used = 0;
#pragma unroll
    for (int k = 0; k < 4; ++k) {
      int be = 0;
      float bv = -1.f;
      for (int e = 0; e < kE; ++e)
        if (!((used >> e) & 1u) && p[e] > bv) { bv = p[e]; be = e; }
      used |= 1u << be;
      te[k] = be;
      tv[k] = bv;
      tsum += bv;
    }
    float r = 1.f / tsum;
#pragma unroll
    for (int k = 0; k < 4; ++k) {
      wts[t * 4 + k]  = tv[k] * r;
      eidx[t * 4 + k] = te[k];
    }
  }
}

// slot assignment (atomics only pick slots; per-row math is order-invariant,
// combine sums in fixed k order -> deterministic output)
__global__ __launch_bounds__(256) void moe_scatter(const int* __restrict__ eidx,
                                                   int* __restrict__ cnt,
                                                   int* __restrict__ idx,
                                                   int* __restrict__ eslot) {
  int t = blockIdx.x * 256 + threadIdx.x;
#pragma unroll
  for (int k = 0; k < 4; ++k) {
    int e = eidx[t * 4 + k];
    int slot = atomicAdd(&cnt[e], 1);
    idx[e * kT + slot] = t;
    eslot[t * 4 + k] = e * kT + slot;
  }
}

__global__ void moe_base(const int* __restrict__ cnt, int* __restrict__ base) {
  if (threadIdx.x == 0) {
    int b = 0;
    for (int e = 0; e < kE; ++e) {
      base[e] = b;
      b += (cnt[e] + 15) & ~15;
    }
  }
}

// ---------------------------------------------------------------------------
// fused gate+up GEMM + clamped swiglu -> act (bf16, compact expert rows)
// block tile M=64 x N=64, K step 32 over H. Double-buffered LDS; A tile via
// per-lane async global->LDS b128 (gathered token rows), W tiles via
// reg-pipelined fp32->bf16 (v_cvt_pk_bf16_f32) stored transposed. 8 waves =
// 4 M x 2 N subtiles, 4 WMMAs per wave per K step.
// ---------------------------------------------------------------------------
__global__ __launch_bounds__(256) void moe_gateup(
    const bf16* __restrict__ xb, const float* __restrict__ wg_all,
    const float* __restrict__ bg_all, const float* __restrict__ wu_all,
    const float* __restrict__ bu_all, const int* __restrict__ cnt,
    const int* __restrict__ idx, const int* __restrict__ base,
    bf16* __restrict__ act) {
  int e = blockIdx.z;
  int ne = cnt[e];
  int npad = (ne + 15) & ~15;
  int m0 = blockIdx.y * 64;
  if (m0 >= npad) return;
  int n0 = blockIdx.x * 64;

  __shared__ __align__(16) bf16 sA[2][64 * 32];
  __shared__ __align__(16) bf16 sG[2][64 * 32];  // transposed [n][k]
  __shared__ __align__(16) bf16 sU[2][64 * 32];  // transposed [n][k]

  int tid = threadIdx.x;
  // A staging: 256 threads x 16B async = 4KB tile
  int ar = tid >> 2, ac = (tid & 3) * 8;
  int slot = m0 + ar;
  int tok = (slot < ne) ? idx[e * kT + slot] : 0;  // clamp pad rows
  const bf16* xrow = xb + (size_t)tok * kH + ac;
  unsigned aoff0 = (unsigned)(uintptr_t)&sA[0][ar * 32 + ac];
  unsigned aoff1 = (unsigned)(uintptr_t)&sA[1][ar * 32 + ac];
  // W staging: thread owns K-pair (2kp,2kp+1) x 4 N columns
  int kp = tid >> 4, nq = (tid & 15) * 4;
  const float* gbase =
      wg_all + (size_t)e * kH * kI + (size_t)(2 * kp) * kI + n0 + nq;
  const float* ubase =
      wu_all + (size_t)e * kH * kI + (size_t)(2 * kp) * kI + n0 + nq;

  int wave = tid >> 5;
  int mt = wave >> 1, nt = wave & 1;
  v8f accg0 = {}, accg1 = {}, accu0 = {}, accu1 = {};
  float4 g0, g1, u0, u1;

  auto fetchW = [&](int k) {
    const float* gp = gbase + (size_t)k * kI;
    const float* up = ubase + (size_t)k * kI;
    __builtin_prefetch(gp + (size_t)64 * kI, 0, 1);
    __builtin_prefetch(up + (size_t)64 * kI, 0, 1);
    g0 = *(const float4*)gp;
    g1 = *(const float4*)(gp + kI);
    u0 = *(const float4*)up;
    u1 = *(const float4*)(up + kI);
  };
  auto stageW = [&](int b) {
    bf16* gt = &sG[b][0];
    bf16* ut = &sU[b][0];
    *(unsigned*)&gt[(nq + 0) * 32 + 2 * kp] = pk2(g0.x, g1.x);
    *(unsigned*)&gt[(nq + 1) * 32 + 2 * kp] = pk2(g0.y, g1.y);
    *(unsigned*)&gt[(nq + 2) * 32 + 2 * kp] = pk2(g0.z, g1.z);
    *(unsigned*)&gt[(nq + 3) * 32 + 2 * kp] = pk2(g0.w, g1.w);
    *(unsigned*)&ut[(nq + 0) * 32 + 2 * kp] = pk2(u0.x, u1.x);
    *(unsigned*)&ut[(nq + 1) * 32 + 2 * kp] = pk2(u0.y, u1.y);
    *(unsigned*)&ut[(nq + 2) * 32 + 2 * kp] = pk2(u0.z, u1.z);
    *(unsigned*)&ut[(nq + 3) * 32 + 2 * kp] = pk2(u0.w, u1.w);
  };

  // prologue: stage K tile 0
  async_copy_b128(aoff0, xrow);
  fetchW(0);
  stageW(0);
  wait_async();

  const int NIT = kH / 32;  // 90
  for (int it = 0; it < NIT; ++it) {
    __syncthreads();
    int cur = it & 1;
    bool more = (it + 1) < NIT;
    if (more) {
      async_copy_b128(cur ? aoff0 : aoff1, xrow + (it + 1) * 32);
      fetchW((it + 1) * 32);
    }
    v16bf fa  = frag_a(&sA[cur][mt * 16 * 32]);
    v16bf fg0 = frag_bt(&sG[cur][(nt * 32) * 32]);
    v16bf fg1 = frag_bt(&sG[cur][(nt * 32 + 16) * 32]);
    v16bf fu0 = frag_bt(&sU[cur][(nt * 32) * 32]);
    v16bf fu1 = frag_bt(&sU[cur][(nt * 32 + 16) * 32]);
    accg0 = __builtin_amdgcn_wmma_f32_16x16x32_bf16(false, fa, false, fg0,
                                                    (short)0, accg0, false, false);
    accg1 = __builtin_amdgcn_wmma_f32_16x16x32_bf16(false, fa, false, fg1,
                                                    (short)0, accg1, false, false);
    accu0 = __builtin_amdgcn_wmma_f32_16x16x32_bf16(false, fa, false, fu0,
                                                    (short)0, accu0, false, false);
    accu1 = __builtin_amdgcn_wmma_f32_16x16x32_bf16(false, fa, false, fu1,
                                                    (short)0, accu1, false, false);
    if (more) {
      stageW(cur ^ 1);
      wait_async();
    }
  }

  // epilogue: bias + gpt-oss clamped swiglu, store bf16 act (two N subtiles)
  int lane = tid & 31;
  int nA = n0 + nt * 32 + (lane & 15);
  int mrow0 = m0 + mt * 16 + ((lane >> 4) << 3);
  float bgA = bg_all[e * kI + nA], bgB = bg_all[e * kI + nA + 16];
  float buA = bu_all[e * kI + nA], buB = bu_all[e * kI + nA + 16];
  size_t rbase = (size_t)base[e];
#pragma unroll
  for (int r = 0; r < 8; ++r) {
    int m = mrow0 + r;
    if (m >= npad) continue;
    bf16* orow = act + (rbase + (size_t)m) * kI;
    float g = fminf(accg0[r] + bgA, kLimit);
    float u = fminf(fmaxf(accu0[r] + buA, -kLimit), kLimit);
    float sg = g * __builtin_amdgcn_rcpf(1.f + __expf(-kAlpha * g));
    orow[nA] = (bf16)((u + 1.f) * sg);
    g = fminf(accg1[r] + bgB, kLimit);
    u = fminf(fmaxf(accu1[r] + buB, -kLimit), kLimit);
    sg = g * __builtin_amdgcn_rcpf(1.f + __expf(-kAlpha * g));
    orow[nA + 16] = (bf16)((u + 1.f) * sg);
  }
}

// ---------------------------------------------------------------------------
// down GEMM: y = act @ w_down[e] + b_down[e]  -> yd (f32, compact rows)
// A tile (64x32 bf16, contiguous rows) staged by the Tensor Data Mover:
// one tensor_load_to_lds per buffer, issued by wave 0 (TDM ignores EXEC),
// waited with s_wait_tensorcnt overlapped behind the WMMAs.
// ---------------------------------------------------------------------------
__global__ __launch_bounds__(256) void moe_down(
    const bf16* __restrict__ act, const float* __restrict__ wd_all,
    const float* __restrict__ bd_all, const int* __restrict__ cnt,
    const int* __restrict__ base, float* __restrict__ yd) {
  int e = blockIdx.z;
  int ne = cnt[e];
  int npad = (ne + 15) & ~15;
  int m0 = blockIdx.y * 64;
  if (m0 >= npad) return;
  int n0 = blockIdx.x * 64;

  __shared__ __align__(16) bf16 sA[2][64 * 32];
  __shared__ __align__(16) bf16 sW[2][64 * 32];  // transposed [n][k]

  int tid = threadIdx.x;
  int wave = tid >> 5;
  size_t rbase = (size_t)base[e];
  int kp = tid >> 4, nq = (tid & 15) * 4;
  const float* wbase =
      wd_all + (size_t)e * kI * kH + (size_t)(2 * kp) * kH + n0 + nq;

#if USE_TDM
  unsigned lds0 = (unsigned)(uintptr_t)&sA[0][0];
  unsigned lds1 = (unsigned)(uintptr_t)&sA[1][0];
  unsigned long long gA =
      (unsigned long long)(uintptr_t)(act + (rbase + (size_t)m0) * kI);
  bool issuer = (wave == 0);
  auto loadA = [&](int buf, int k) {
    if (issuer) tdm_load_2d(buf ? lds1 : lds0, gA + (unsigned long long)k * 2, kI);
  };
  auto waitA = [&]() {
    if (issuer) wait_tensor();
  };
#else
  int ar = tid >> 2, ac = (tid & 3) * 8;
  const bf16* arow = act + (rbase + (size_t)(m0 + ar)) * kI + ac;
  unsigned aoff0 = (unsigned)(uintptr_t)&sA[0][ar * 32 + ac];
  unsigned aoff1 = (unsigned)(uintptr_t)&sA[1][ar * 32 + ac];
  auto loadA = [&](int buf, int k) {
    async_copy_b128(buf ? aoff1 : aoff0, arow + k);
  };
  auto waitA = [&]() { wait_async(); };
#endif

  int mt = wave >> 1, nt = wave & 1;
  v8f acc0 = {}, acc1 = {};
  float4 w0, w1;

  auto fetchW = [&](int k) {
    const float* wp = wbase + (size_t)k * kH;
    __builtin_prefetch(wp + (size_t)64 * kH, 0, 1);
    w0 = *(const float4*)wp;
    w1 = *(const float4*)(wp + kH);
  };
  auto stageW = [&](int b) {
    bf16* wt = &sW[b][0];
    *(unsigned*)&wt[(nq + 0) * 32 + 2 * kp] = pk2(w0.x, w1.x);
    *(unsigned*)&wt[(nq + 1) * 32 + 2 * kp] = pk2(w0.y, w1.y);
    *(unsigned*)&wt[(nq + 2) * 32 + 2 * kp] = pk2(w0.z, w1.z);
    *(unsigned*)&wt[(nq + 3) * 32 + 2 * kp] = pk2(w0.w, w1.w);
  };

  loadA(0, 0);
  fetchW(0);
  stageW(0);
  waitA();

  const int NIT = kI / 32;  // 90
  for (int it = 0; it < NIT; ++it) {
    __syncthreads();
    int cur = it & 1;
    bool more = (it + 1) < NIT;
    if (more) {
      loadA(cur ^ 1, (it + 1) * 32);
      fetchW((it + 1) * 32);
    }
    v16bf fa  = frag_a(&sA[cur][mt * 16 * 32]);
    v16bf fb0 = frag_bt(&sW[cur][(nt * 32) * 32]);
    v16bf fb1 = frag_bt(&sW[cur][(nt * 32 + 16) * 32]);
    acc0 = __builtin_amdgcn_wmma_f32_16x16x32_bf16(false, fa, false, fb0,
                                                   (short)0, acc0, false, false);
    acc1 = __builtin_amdgcn_wmma_f32_16x16x32_bf16(false, fa, false, fb1,
                                                   (short)0, acc1, false, false);
    if (more) {
      stageW(cur ^ 1);
      waitA();
    }
  }

  int lane = tid & 31;
  int n = n0 + nt * 32 + (lane & 15);
  int mrow0 = m0 + mt * 16 + ((lane >> 4) << 3);
  float bdA = bd_all[e * kH + n], bdB = bd_all[e * kH + n + 16];
#pragma unroll
  for (int r = 0; r < 8; ++r) {
    int m = mrow0 + r;
    if (m >= npad) continue;
    float* yrow = yd + (rbase + (size_t)m) * kH;
    yrow[n]      = acc0[r] + bdA;
    yrow[n + 16] = acc1[r] + bdB;
  }
}

// deterministic combine: each token sums its 4 expert rows in fixed k order
__global__ __launch_bounds__(288) void moe_combine(
    const float* __restrict__ yd, const float* __restrict__ wts,
    const int* __restrict__ eslot, const int* __restrict__ base,
    float* __restrict__ out) {
  int t = blockIdx.y;
  int h = blockIdx.x * 288 + threadIdx.x;
  float s = 0.f;
#pragma unroll
  for (int k = 0; k < 4; ++k) {
    int es = eslot[t * 4 + k];
    int row = base[es >> 10] + (es & 1023);  // kT == 1024
    s += wts[t * 4 + k] * yd[(size_t)row * kH + h];
  }
  out[(size_t)t * kH + h] = s;
}

// ---------------------------------------------------------------------------
extern "C" void kernel_launch(void* const* d_in, const int* in_sizes, int n_in,
                              void* d_out, int out_size, void* d_ws,
                              size_t ws_size, hipStream_t stream) {
  (void)in_sizes; (void)n_in; (void)out_size; (void)ws_size;
  const float* x  = (const float*)d_in[0];
  const float* rw = (const float*)d_in[1];
  const float* rb = (const float*)d_in[2];
  const float* wg = (const float*)d_in[3];
  const float* bg = (const float*)d_in[4];
  const float* wu = (const float*)d_in[5];
  const float* bu = (const float*)d_in[6];
  const float* wd = (const float*)d_in[7];
  const float* bd = (const float*)d_in[8];
  float* out = (float*)d_out;

  char* p = (char*)d_ws;
  size_t off = 0;
  auto take = [&](size_t bytes) -> void* {
    void* q = p + off;
    off = (off + bytes + 255) & ~(size_t)255;
    return q;
  };
  bf16*  xb    = (bf16*)take((size_t)kT * kH * sizeof(bf16));
  float* wts   = (float*)take((size_t)kT * 4 * sizeof(float));
  int*   eidx  = (int*)take((size_t)kT * 4 * sizeof(int));
  int*   eslot = (int*)take((size_t)kT * 4 * sizeof(int));
  int*   cnt   = (int*)take(kE * sizeof(int));
  int*   base  = (int*)take(kE * sizeof(int));
  int*   idx   = (int*)take((size_t)kE * kT * sizeof(int));
  bf16*  act   = (bf16*)take((size_t)kRowsMax * kI * sizeof(bf16));
  float* yd    = (float*)take((size_t)kRowsMax * kH * sizeof(float));

  moe_init<<<1, 32, 0, stream>>>(cnt);
  moe_cvt<<<(kT * kH) / 1024, 256, 0, stream>>>(x, xb);
  moe_router<<<kT / 8, 256, 0, stream>>>(x, rw, rb, wts, eidx);
  moe_scatter<<<kT / 256, 256, 0, stream>>>(eidx, cnt, idx, eslot);
  moe_base<<<1, 1, 0, stream>>>(cnt, base);

  dim3 g1(kI / 64, kT / 64, kE);
  moe_gateup<<<g1, 256, 0, stream>>>(xb, wg, bg, wu, bu, cnt, idx, base, act);
  dim3 g2(kH / 64, kT / 64, kE);
  moe_down<<<g2, 256, 0, stream>>>(act, wd, bd, cnt, base, yd);
  dim3 g3(kH / 288, kT);
  moe_combine<<<g3, 288, 0, stream>>>(yd, wts, eslot, base, out);
}
